// MSARowAttentionWithBias_37417755083154
// MI455X (gfx1250) — compile-verified, matrix operand order checked by reading, not run on hardware
//
#include <hip/hip_runtime.h>
#include <hip/hip_bf16.h>
#include <math.h>

// ---------------------------------------------------------------------------
// MSARowAttentionWithBias for MI455X (gfx1250).
// B=1, N=256 (s), L=384, D_MSA=256, D_PAIR=128, H=8, DH=32, HD=256.
// All contractions via v_wmma_f32_16x16x32_bf16 (wave32 WMMA), fragments as
// global_load_b128, copy-free ping-pong K loops to overlap loads with WMMA.
// ---------------------------------------------------------------------------

typedef __attribute__((ext_vector_type(16))) __bf16 v16bf;
typedef __attribute__((ext_vector_type(8)))  float  v8f;

#define DEV static __device__ __forceinline__

DEV unsigned short f2bf(float f) {
  unsigned u = __float_as_uint(f);
  unsigned r = u + 0x7FFFu + ((u >> 16) & 1u);   // round-to-nearest-even
  return (unsigned short)(r >> 16);
}
DEV float bf2f(unsigned short h) { return __uint_as_float(((unsigned)h) << 16); }

union BFrag { v16bf v; unsigned u[8]; };
union CFrag { v8f  v; float   f[8]; };

// K index pattern for 16-bit A/B fragments (CDNA5 ISA 7.12.2):
// lanes 0-15: regs 0-3 -> K pairs {0,1},{2,3},{4,5},{6,7}; regs 4-7 -> {16,17}..{22,23}
// lanes 16-31: +8.
DEV int frag_k(int r, int lane) {
  int k = (r < 4) ? (2 * r) : (16 + 2 * (r - 4));
  return k + ((lane & 16) ? 8 : 0);
}

// Fragment where element (row, k) lives at p[row*ld + k] (k contiguous).
// Works for A (row=M) and for B when stored n-major (row=N, i.e. B^T row-major).
// Per lane this is two contiguous 16B runs -> compiler emits global_load_b128.
DEV void frag_load_rm(BFrag& f, const unsigned short* __restrict__ p,
                      int ld, int row0, int k0) {
  int lane = threadIdx.x & 31;
  const unsigned short* rp = p + (row0 + (lane & 15)) * ld + k0;
#pragma unroll
  for (int r = 0; r < 8; ++r) {
    int k = frag_k(r, lane);
    f.u[r] = *(const unsigned*)(rp + k);        // 2 packed bf16
  }
}

DEV v8f wmma_bf(const BFrag& a, const BFrag& b, v8f c) {
  return __builtin_amdgcn_wmma_f32_16x16x32_bf16(false, a.v, false, b.v,
                                                 (short)0, c, false, false);
}

// ---------------------------------------------------------------------------
// 1) Weight convert + transpose: wT[slot][n*256+k] = bf16(W[k*256+n]); 7 slots.
// slots: 0=sw_q_w 1=sw_k_w 2=wq 3=wk 4=wv 5=wg 6=wo
// ---------------------------------------------------------------------------
__global__ void k_convw(const float* __restrict__ w0, const float* __restrict__ w1,
                        const float* __restrict__ w2, const float* __restrict__ w3,
                        const float* __restrict__ w4, const float* __restrict__ w5,
                        const float* __restrict__ w6, unsigned short* __restrict__ wT) {
  int idx = blockIdx.x * 256 + threadIdx.x;       // 7*65536 threads
  int slot = idx >> 16, e = idx & 65535;
  int k = e >> 8, n = e & 255;
  const float* w = slot == 0 ? w0 : slot == 1 ? w1 : slot == 2 ? w2 :
                   slot == 3 ? w3 : slot == 4 ? w4 : slot == 5 ? w5 : w6;
  wT[(size_t)slot * 65536 + n * 256 + k] = f2bf(w[k * 256 + n]);
}

// ---------------------------------------------------------------------------
// 2) LayerNorm over 256 features, one wave per token, bf16 out.
// ---------------------------------------------------------------------------
__global__ void k_ln_msa(const float* __restrict__ x, const float* __restrict__ g,
                         const float* __restrict__ b, unsigned short* __restrict__ o) {
  int wave = threadIdx.x >> 5, lane = threadIdx.x & 31;
  int tok = blockIdx.x * 8 + wave;                // 98304 tokens
  const float* xr = x + (size_t)tok * 256;
  float v[8]; float s = 0.f;
#pragma unroll
  for (int j = 0; j < 8; ++j) { v[j] = xr[lane + 32 * j]; s += v[j]; }
#pragma unroll
  for (int m = 16; m; m >>= 1) s += __shfl_xor(s, m, 32);
  float mean = s * (1.f / 256.f);
  float q = 0.f;
#pragma unroll
  for (int j = 0; j < 8; ++j) { float d = v[j] - mean; q += d * d; }
#pragma unroll
  for (int m = 16; m; m >>= 1) q += __shfl_xor(q, m, 32);
  float inv = rsqrtf(q * (1.f / 256.f) + 1e-5f);
#pragma unroll
  for (int j = 0; j < 8; ++j) {
    int fidx = lane + 32 * j;
    o[(size_t)tok * 256 + fidx] = f2bf((v[j] - mean) * inv * g[fidx] + b[fidx]);
  }
}

// LayerNorm over 128 features (pair), one wave per (i,j) cell.
__global__ void k_ln_pair(const float* __restrict__ x, const float* __restrict__ g,
                          const float* __restrict__ b, unsigned short* __restrict__ o) {
  int wave = threadIdx.x >> 5, lane = threadIdx.x & 31;
  int tok = blockIdx.x * 8 + wave;                // 147456 cells
  const float* xr = x + (size_t)tok * 128;
  float v[4]; float s = 0.f;
#pragma unroll
  for (int j = 0; j < 4; ++j) { v[j] = xr[lane + 32 * j]; s += v[j]; }
#pragma unroll
  for (int m = 16; m; m >>= 1) s += __shfl_xor(s, m, 32);
  float mean = s * (1.f / 128.f);
  float q = 0.f;
#pragma unroll
  for (int j = 0; j < 4; ++j) { float d = v[j] - mean; q += d * d; }
#pragma unroll
  for (int m = 16; m; m >>= 1) q += __shfl_xor(q, m, 32);
  float inv = rsqrtf(q * (1.f / 128.f) + 1e-5f);
#pragma unroll
  for (int j = 0; j < 4; ++j) {
    int fidx = lane + 32 * j;
    o[(size_t)tok * 128 + fidx] = f2bf((v[j] - mean) * inv * g[fidx] + b[fidx]);
  }
}

// ---------------------------------------------------------------------------
// 3) Fused 5-way projection GEMM: msa_n(98304x256) @ {sw_k,wq,wk,wv,wg}.
// Wave tile 16x64, copy-free ping-pong over 8 K-steps. grid(768, 20).
// ---------------------------------------------------------------------------
__global__ void __launch_bounds__(256)
k_gemm_proj(const unsigned short* __restrict__ A, const unsigned short* __restrict__ wT,
            unsigned short* __restrict__ swkb, unsigned short* __restrict__ qb,
            unsigned short* __restrict__ kb,   unsigned short* __restrict__ vb,
            unsigned short* __restrict__ gb,
            const float* __restrict__ sw_k_b, const float* __restrict__ bg) {
  int wave = threadIdx.x >> 5, lane = threadIdx.x & 31;
  int row0 = (blockIdx.x * 8 + wave) * 16;
  int mg = blockIdx.y, mat = mg >> 2, n0 = (mg & 3) * 64;
  const unsigned short* B = wT + (size_t)(mat + 1) * 65536;   // slots 1..5
  CFrag c[4];
#pragma unroll
  for (int t = 0; t < 4; ++t)
#pragma unroll
    for (int r = 0; r < 8; ++r) c[t].f[r] = 0.f;
  BFrag a[2], b[2][4];
  frag_load_rm(a[0], A, 256, row0, 0);
#pragma unroll
  for (int t = 0; t < 4; ++t) frag_load_rm(b[0][t], B, 256, n0 + 16 * t, 0);
#pragma unroll
  for (int i = 0; i < 7; ++i) {                   // 8 K-steps, ping-pong
    int cur = i & 1, nxt = cur ^ 1, kk = 32 * (i + 1);
    frag_load_rm(a[nxt], A, 256, row0, kk);
#pragma unroll
    for (int t = 0; t < 4; ++t) frag_load_rm(b[nxt][t], B, 256, n0 + 16 * t, kk);
#pragma unroll
    for (int t = 0; t < 4; ++t) c[t].v = wmma_bf(a[cur], b[cur][t], c[t].v);
  }
#pragma unroll
  for (int t = 0; t < 4; ++t) c[t].v = wmma_bf(a[1], b[1][t], c[t].v);

  unsigned short* out = mat == 0 ? swkb : mat == 1 ? qb : mat == 2 ? kb :
                        mat == 3 ? vb : gb;
#pragma unroll
  for (int t = 0; t < 4; ++t)
#pragma unroll
    for (int r = 0; r < 8; ++r) {
      int m = row0 + r + ((lane & 16) ? 8 : 0);
      int col = n0 + 16 * t + (lane & 15);
      float val = c[t].f[r];
      if (mat == 0)      val += sw_k_b[col];
      else if (mat == 2) val *= 0.1767766953f;                 // 1/sqrt(32)
      else if (mat == 4) val = 1.f / (1.f + __expf(-(val + bg[col])));
      out[(size_t)m * 256 + col] = f2bf(val);
    }
}

// ---------------------------------------------------------------------------
// 4) swq = (tar @ sw_q_w + b) * scale ; tar = rows 0..383 of msa_n (n==0).
// ---------------------------------------------------------------------------
__global__ void __launch_bounds__(256)
k_gemm_swq(const unsigned short* __restrict__ A, const unsigned short* __restrict__ wT,
           const float* __restrict__ sw_q_b, unsigned short* __restrict__ swq) {
  int wave = threadIdx.x >> 5, lane = threadIdx.x & 31;
  int row0 = (blockIdx.x * 8 + wave) * 16;       // < 384
  int n0 = blockIdx.y * 64;
  CFrag c[4];
#pragma unroll
  for (int t = 0; t < 4; ++t)
#pragma unroll
    for (int r = 0; r < 8; ++r) c[t].f[r] = 0.f;
  BFrag a[2], b[2][4];
  frag_load_rm(a[0], A, 256, row0, 0);
#pragma unroll
  for (int t = 0; t < 4; ++t) frag_load_rm(b[0][t], wT, 256, n0 + 16 * t, 0);
#pragma unroll
  for (int i = 0; i < 7; ++i) {
    int cur = i & 1, nxt = cur ^ 1, kk = 32 * (i + 1);
    frag_load_rm(a[nxt], A, 256, row0, kk);
#pragma unroll
    for (int t = 0; t < 4; ++t) frag_load_rm(b[nxt][t], wT, 256, n0 + 16 * t, kk);
#pragma unroll
    for (int t = 0; t < 4; ++t) c[t].v = wmma_bf(a[cur], b[cur][t], c[t].v);
  }
#pragma unroll
  for (int t = 0; t < 4; ++t) c[t].v = wmma_bf(a[1], b[1][t], c[t].v);
#pragma unroll
  for (int t = 0; t < 4; ++t)
#pragma unroll
    for (int r = 0; r < 8; ++r) {
      int m = row0 + r + ((lane & 16) ? 8 : 0);
      int col = n0 + 16 * t + (lane & 15);
      swq[(size_t)m * 256 + col] = f2bf((c[t].f[r] + sw_q_b[col]) * 0.1767766953f);
    }
}

// ---------------------------------------------------------------------------
// 5) seq_weight[n,l,h] = softmax_n( sum_d swq[l,h,d]*swk[n,l,h,d] ).
// ---------------------------------------------------------------------------
__global__ void k_seqw(const unsigned short* __restrict__ swq,
                       const unsigned short* __restrict__ swk,
                       float* __restrict__ seqw) {
  int l = blockIdx.x, h = blockIdx.y, t = threadIdx.x;
  __shared__ float sq[32];
  __shared__ float red[256];
  if (t < 32) sq[t] = bf2f(swq[l * 256 + h * 32 + t]);
  __syncthreads();
  const unsigned short* kr = swk + ((size_t)(t * 384 + l) * 8 + h) * 32;
  float s = 0.f;
#pragma unroll
  for (int d = 0; d < 32; ++d) s += bf2f(kr[d]) * sq[d];
  red[t] = s; __syncthreads();
  for (int m = 128; m; m >>= 1) { if (t < m) red[t] = fmaxf(red[t], red[t + m]); __syncthreads(); }
  float mx = red[0]; __syncthreads();
  float e = __expf(s - mx);
  red[t] = e; __syncthreads();
  for (int m = 128; m; m >>= 1) { if (t < m) red[t] += red[t + m]; __syncthreads(); }
  seqw[(size_t)(t * 384 + l) * 8 + h] = e / red[0];
}

// 6) query *= seq_weight (in-place on qb).
__global__ void k_scale_q(unsigned short* __restrict__ qb,
                          const float* __restrict__ seqw) {
  int i = blockIdx.x * 256 + threadIdx.x;         // 25165824
  qb[i] = f2bf(bf2f(qb[i]) * seqw[i >> 5]);
}

// 7) bias[q,k,h] = pair_n[q,k,:] . wb[:,h]   (128-dot, VALU — tiny FLOPs)
__global__ void k_bias(const unsigned short* __restrict__ pn,
                       const float* __restrict__ wb, float* __restrict__ biasb) {
  int t = blockIdx.x * 256 + threadIdx.x;         // 147456*8
  int p = t >> 3, h = t & 7;
  const unsigned short* pr = pn + (size_t)p * 128;
  float s = 0.f;
  for (int c = 0; c < 128; ++c) s += bf2f(pr[c]) * wb[c * 8 + h];
  biasb[t] = s;
}

// ---------------------------------------------------------------------------
// 8) attn[q,k,h] = sum_{s,d} q_s[s,q,h,d]*key[s,k,h,d] + bias.
// s-loop software pipeline, both phases written out (no buffer copies).
// grid(3,6,8).
// ---------------------------------------------------------------------------
__global__ void __launch_bounds__(256)
k_attn_gemm(const unsigned short* __restrict__ qb, const unsigned short* __restrict__ kb,
            const float* __restrict__ biasb, float* __restrict__ attnf) {
  int wave = threadIdx.x >> 5, lane = threadIdx.x & 31;
  int q0 = (blockIdx.x * 8 + wave) * 16;
  int k0 = blockIdx.y * 64;
  int h = blockIdx.z;
  CFrag c[4];
#pragma unroll
  for (int t = 0; t < 4; ++t)
#pragma unroll
    for (int r = 0; r < 8; ++r) c[t].f[r] = 0.f;
  const unsigned short* Ab = qb + h * 32;         // + s*98304, [q][256]
  const unsigned short* Bb = kb + h * 32;
  BFrag a[2], b[2][4];
  frag_load_rm(a[0], Ab, 256, q0, 0);
#pragma unroll
  for (int t = 0; t < 4; ++t) frag_load_rm(b[0][t], Bb, 256, k0 + 16 * t, 0);
  for (int s = 0; s < 254; s += 2) {
    const unsigned short* A1 = Ab + (size_t)(s + 1) * 98304;
    const unsigned short* B1 = Bb + (size_t)(s + 1) * 98304;
    frag_load_rm(a[1], A1, 256, q0, 0);
#pragma unroll
    for (int t = 0; t < 4; ++t) frag_load_rm(b[1][t], B1, 256, k0 + 16 * t, 0);
#pragma unroll
    for (int t = 0; t < 4; ++t) c[t].v = wmma_bf(a[0], b[0][t], c[t].v);
    const unsigned short* A2 = Ab + (size_t)(s + 2) * 98304;
    const unsigned short* B2 = Bb + (size_t)(s + 2) * 98304;
    frag_load_rm(a[0], A2, 256, q0, 0);
#pragma unroll
    for (int t = 0; t < 4; ++t) frag_load_rm(b[0][t], B2, 256, k0 + 16 * t, 0);
#pragma unroll
    for (int t = 0; t < 4; ++t) c[t].v = wmma_bf(a[1], b[1][t], c[t].v);
  }
  {
    const unsigned short* A1 = Ab + (size_t)255 * 98304;
    const unsigned short* B1 = Bb + (size_t)255 * 98304;
    frag_load_rm(a[1], A1, 256, q0, 0);
#pragma unroll
    for (int t = 0; t < 4; ++t) frag_load_rm(b[1][t], B1, 256, k0 + 16 * t, 0);
#pragma unroll
    for (int t = 0; t < 4; ++t) c[t].v = wmma_bf(a[0], b[0][t], c[t].v);
#pragma unroll
    for (int t = 0; t < 4; ++t) c[t].v = wmma_bf(a[1], b[1][t], c[t].v);
  }
#pragma unroll
  for (int t = 0; t < 4; ++t)
#pragma unroll
    for (int r = 0; r < 8; ++r) {
      int q = q0 + r + ((lane & 16) ? 8 : 0);
      int kc = k0 + 16 * t + (lane & 15);
      attnf[((size_t)h * 384 + q) * 384 + kc] =
          c[t].f[r] + biasb[((size_t)q * 384 + kc) * 8 + h];
    }
}

// 9) softmax over k per (h,q) row of 384; store bf16 for the AV GEMM.
__global__ void k_attn_softmax(const float* __restrict__ attnf,
                               unsigned short* __restrict__ attnb) {
  int row = blockIdx.x;                           // h*384+q, 3072 rows
  const float* r = attnf + (size_t)row * 384;
  __shared__ float red[128];
  int t = threadIdx.x;
  float v0 = r[t], v1 = r[t + 128], v2 = r[t + 256];
  red[t] = fmaxf(fmaxf(v0, v1), v2); __syncthreads();
  for (int m = 64; m; m >>= 1) { if (t < m) red[t] = fmaxf(red[t], red[t + m]); __syncthreads(); }
  float mx = red[0]; __syncthreads();
  float e0 = __expf(v0 - mx), e1 = __expf(v1 - mx), e2 = __expf(v2 - mx);
  red[t] = e0 + e1 + e2; __syncthreads();
  for (int m = 64; m; m >>= 1) { if (t < m) red[t] += red[t + m]; __syncthreads(); }
  float inv = 1.f / red[0];
  unsigned short* o = attnb + (size_t)row * 384;
  o[t] = f2bf(e0 * inv); o[t + 128] = f2bf(e1 * inv); o[t + 256] = f2bf(e2 * inv);
}

// ---------------------------------------------------------------------------
// 10) V transpose: vT[s][h][d][k] = v[s][k][h][d]. LDS tile per (s,h),
// padded stride 386 (odd word step) -> bank-conflict-free scatter.
// ---------------------------------------------------------------------------
__global__ void __launch_bounds__(256)
k_vt(const unsigned short* __restrict__ vb, unsigned short* __restrict__ vt) {
  __shared__ unsigned short tile[32 * 386];
  int s = blockIdx.x, h = blockIdx.y, t = threadIdx.x;
  const unsigned short* src = vb + (size_t)s * 98304 + h * 32;
#pragma unroll
  for (int j = 0; j < 48; ++j) {                  // 384*32 / 256
    int idx = t + 256 * j;                        // d fastest -> coalesced reads
    int k = idx >> 5, d = idx & 31;
    tile[d * 386 + k] = src[k * 256 + d];
  }
  __syncthreads();
  unsigned short* dst = vt + ((size_t)(s * 8 + h) * 32) * 384;
#pragma unroll
  for (int j = 0; j < 48; ++j) {
    int idx = t + 256 * j;                        // k fastest -> coalesced writes
    int d = idx / 384, k = idx - d * 384;
    dst[d * 384 + k] = tile[d * 386 + k];
  }
}

// ---------------------------------------------------------------------------
// 11) out[s,q,h,d] = sum_k attn[h,q,k]*v[s,k,h,d]; fused gate multiply.
// grid(256, 3, 8): wave tile 16q x 32d, 12 K-steps fully unrolled ping-pong.
// ---------------------------------------------------------------------------
__global__ void __launch_bounds__(256)
k_out_gemm(const unsigned short* __restrict__ attnb, const unsigned short* __restrict__ vt,
           const unsigned short* __restrict__ gateb, unsigned short* __restrict__ og) {
  int wave = threadIdx.x >> 5, lane = threadIdx.x & 31;
  int s = blockIdx.x;
  int q0 = (blockIdx.y * 8 + wave) * 16;
  int h = blockIdx.z;
  const unsigned short* Ab = attnb + (size_t)h * 384 * 384;         // [q][384]
  const unsigned short* Bb = vt + ((size_t)(s * 8 + h) * 32) * 384; // [d][384]
  CFrag c[2];
#pragma unroll
  for (int t = 0; t < 2; ++t)
#pragma unroll
    for (int r = 0; r < 8; ++r) c[t].f[r] = 0.f;
  BFrag a[2], b[2][2];
  frag_load_rm(a[0], Ab, 384, q0, 0);
#pragma unroll
  for (int t = 0; t < 2; ++t) frag_load_rm(b[0][t], Bb, 384, 16 * t, 0);
#pragma unroll
  for (int i = 0; i < 11; ++i) {                  // 12 K-steps, ping-pong
    int cur = i & 1, nxt = cur ^ 1, kk = 32 * (i + 1);
    frag_load_rm(a[nxt], Ab, 384, q0, kk);
#pragma unroll
    for (int t = 0; t < 2; ++t) frag_load_rm(b[nxt][t], Bb, 384, 16 * t, kk);
#pragma unroll
    for (int t = 0; t < 2; ++t) c[t].v = wmma_bf(a[cur], b[cur][t], c[t].v);
  }
#pragma unroll
  for (int t = 0; t < 2; ++t) c[t].v = wmma_bf(a[1], b[1][t], c[t].v);
#pragma unroll
  for (int t = 0; t < 2; ++t)
#pragma unroll
    for (int r = 0; r < 8; ++r) {
      int q = q0 + r + ((lane & 16) ? 8 : 0);
      int col = h * 32 + 16 * t + (lane & 15);
      size_t oi = ((size_t)s * 384 + q) * 256 + col;
      og[oi] = f2bf(c[t].f[r] * bf2f(gateb[oi]));
    }
}

// 12) final: d_out = og(98304x256) @ wo + bo  (fp32 out). grid(768,4).
__global__ void __launch_bounds__(256)
k_gemm_out(const unsigned short* __restrict__ A, const unsigned short* __restrict__ wT,
           const float* __restrict__ bo, float* __restrict__ out) {
  int wave = threadIdx.x >> 5, lane = threadIdx.x & 31;
  int row0 = (blockIdx.x * 8 + wave) * 16;
  int n0 = blockIdx.y * 64;
  const unsigned short* B = wT + (size_t)6 * 65536;             // slot 6 = wo
  CFrag c[4];
#pragma unroll
  for (int t = 0; t < 4; ++t)
#pragma unroll
    for (int r = 0; r < 8; ++r) c[t].f[r] = 0.f;
  BFrag a[2], b[2][4];
  frag_load_rm(a[0], A, 256, row0, 0);
#pragma unroll
  for (int t = 0; t < 4; ++t) frag_load_rm(b[0][t], B, 256, n0 + 16 * t, 0);
#pragma unroll
  for (int i = 0; i < 7; ++i) {
    int cur = i & 1, nxt = cur ^ 1, kk = 32 * (i + 1);
    frag_load_rm(a[nxt], A, 256, row0, kk);
#pragma unroll
    for (int t = 0; t < 4; ++t) frag_load_rm(b[nxt][t], B, 256, n0 + 16 * t, kk);
#pragma unroll
    for (int t = 0; t < 4; ++t) c[t].v = wmma_bf(a[cur], b[cur][t], c[t].v);
  }
#pragma unroll
  for (int t = 0; t < 4; ++t) c[t].v = wmma_bf(a[1], b[1][t], c[t].v);
#pragma unroll
  for (int t = 0; t < 4; ++t)
#pragma unroll
    for (int r = 0; r < 8; ++r) {
      int m = row0 + r + ((lane & 16) ? 8 : 0);
      int col = n0 + 16 * t + (lane & 15);
      out[(size_t)m * 256 + col] = c[t].f[r] + bo[col];
    }
}

// ---------------------------------------------------------------------------
extern "C" void kernel_launch(void* const* d_in, const int* in_sizes, int n_in,
                              void* d_out, int out_size, void* d_ws, size_t ws_size,
                              hipStream_t stream) {
  const float* msa    = (const float*)d_in[0];
  const float* pair   = (const float*)d_in[1];
  const float* lmg    = (const float*)d_in[2];
  const float* lmb    = (const float*)d_in[3];
  const float* lpg    = (const float*)d_in[4];
  const float* lpb    = (const float*)d_in[5];
  const float* sw_q_w = (const float*)d_in[6];
  const float* sw_q_b = (const float*)d_in[7];
  const float* sw_k_w = (const float*)d_in[8];
  const float* sw_k_b = (const float*)d_in[9];
  const float* wq     = (const float*)d_in[10];
  const float* wk     = (const float*)d_in[11];
  const float* wv     = (const float*)d_in[12];
  const float* wb     = (const float*)d_in[13];
  const float* wg     = (const float*)d_in[14];
  const float* bg     = (const float*)d_in[15];
  const float* wo     = (const float*)d_in[16];
  const float* bo     = (const float*)d_in[17];

  char* p = (char*)d_ws;
  auto alloc = [&](size_t bytes) {
    char* r = p; p += (bytes + 255) & ~(size_t)255; return r;
  };
  unsigned short* msa_n  = (unsigned short*)alloc(98304ull * 256 * 2);
  unsigned short* pair_n = (unsigned short*)alloc(147456ull * 128 * 2);
  unsigned short* wT     = (unsigned short*)alloc(7ull * 65536 * 2);
  unsigned short* swkb   = (unsigned short*)alloc(98304ull * 256 * 2);
  unsigned short* qb     = (unsigned short*)alloc(98304ull * 256 * 2);
  unsigned short* kb     = (unsigned short*)alloc(98304ull * 256 * 2);
  unsigned short* vb     = (unsigned short*)alloc(98304ull * 256 * 2);
  unsigned short* gateb  = (unsigned short*)alloc(98304ull * 256 * 2);
  unsigned short* vtb    = (unsigned short*)alloc(98304ull * 256 * 2);
  unsigned short* swqb   = (unsigned short*)alloc(384ull * 256 * 2);
  float*          seqw   = (float*)alloc(256ull * 384 * 8 * 4);
  float*          biasb  = (float*)alloc(147456ull * 8 * 4);
  float*          attnf  = (float*)alloc(8ull * 384 * 384 * 4);
  unsigned short* attnb  = (unsigned short*)alloc(8ull * 384 * 384 * 2);
  unsigned short* og     = (unsigned short*)alloc(98304ull * 256 * 2);

  k_convw<<<1792, 256, 0, stream>>>(sw_q_w, sw_k_w, wq, wk, wv, wg, wo, wT);
  k_ln_msa<<<12288, 256, 0, stream>>>(msa, lmg, lmb, msa_n);
  k_ln_pair<<<18432, 256, 0, stream>>>(pair, lpg, lpb, pair_n);
  k_gemm_proj<<<dim3(768, 20), 256, 0, stream>>>(msa_n, wT, swkb, qb, kb, vb,
                                                 gateb, sw_k_b, bg);
  k_gemm_swq<<<dim3(3, 4), 256, 0, stream>>>(msa_n, wT, sw_q_b, swqb);
  k_seqw<<<dim3(384, 8), 256, 0, stream>>>(swqb, swkb, seqw);
  k_scale_q<<<98304, 256, 0, stream>>>(qb, seqw);
  k_bias<<<4608, 256, 0, stream>>>(pair_n, wb, biasb);
  k_vt<<<dim3(256, 8), 256, 0, stream>>>(vb, vtb);
  k_attn_gemm<<<dim3(3, 6, 8), 256, 0, stream>>>(qb, kb, biasb, attnf);
  k_attn_softmax<<<3072, 128, 0, stream>>>(attnf, attnb);
  k_out_gemm<<<dim3(256, 3, 8), 256, 0, stream>>>(attnb, vtb, gateb, og);
  k_gemm_out<<<dim3(768, 4), 256, 0, stream>>>(og, wT, bo, (float*)d_out);
  (void)in_sizes; (void)n_in; (void)out_size; (void)ws_size;
}
